// DispersionFilter_5128190951836
// MI455X (gfx1250) — matrix operand order repaired and checked by reading, not compiled, not run on hardware
//
#include <hip/hip_runtime.h>

// CDNA5 / gfx1250: wave32, WMMA 16x16x32 f16 -> f32 accumulate.
typedef __attribute__((ext_vector_type(16))) _Float16 v16h;
typedef __attribute__((ext_vector_type(8)))  float    v8f;
typedef __attribute__((ext_vector_type(4)))  float    v4f;

#define STIFF      0.05f
#define LOG2_STIFF -4.321928094887362f   // log2(0.05)

// Problem shape from the reference (compile-time so tile->row is shift/mask).
#define T_LEN      262144
#define TPR_LOG2   10                    // tilesPerRow = T/256 = 1024
#define TPR_MASK   1023

// One wave computes 256 consecutive outputs of one row as
//   Y(16x16) = W(16x32) x X(32x16)   (16-sample halo; s^16 ~ 1.5e-21 << f32 eps)
// with hi/lo f16 splitting for f32-grade accuracy (3 chained WMMAs).
// blockIdx.y selects the channel (left/right) so pointer selection is
// SGPR-uniform; the tile index is readfirstlane'd so all row/base addressing
// stays scalar and loads/stores use the SADDR form.
__global__ __launch_bounds__(256)
void dispersion_iir_wmma(const float* __restrict__ left,
                         const float* __restrict__ right,
                         float* __restrict__ out,
                         const int R) {
  const int lane = threadIdx.x & 31;
  const int hi   = (lane >= 16) ? 1 : 0;   // high half of the wave
  const int M    = lane & 15;              // row of A / position-in-block

  // ---- Build FIR weight matrix A = Whi + Wlo (f16 split), once per thread.
  // A[m][k] = (1-s) * s^(m + 16 - k)  for k <= m+16, else 0.
  v16h ahi, alo;
#pragma unroll
  for (int e = 0; e < 16; ++e) {
    const int K = e + (hi ? 16 : 0);  // contiguous k order, matches B
    const int d = M + 16 - K;                          // tap depth in -15..31
    float w = (d >= 0) ? (1.0f - STIFF) * exp2f((float)d * LOG2_STIFF) : 0.0f;
    const _Float16 whi = (_Float16)w;
    ahi[e] = whi;
    alo[e] = (_Float16)(w - (float)whi);
  }

  // Channel is uniform per block (SGPR select, no exec-mask divergence).
  const float* chan = blockIdx.y ? right : left;
  float* outc = out + (size_t)blockIdx.y * (size_t)R * T_LEN;

  // Per-lane loop-invariant offsets (elements).
  const int bOff = M * 16 - 16 + hi * 16;  // B-load offset within tile (halo)
  const int oOff = M * 16 + hi * 8;        // store offset within tile

  const int nTiles      = R << TPR_LOG2;   // tiles in this channel
  const int wavesPerBlk = blockDim.x >> 5;
  const int nWaves      = gridDim.x * wavesPerBlk;
  // Wave-uniform by construction; make it scalar for SALU loop control.
  const int wave0 = __builtin_amdgcn_readfirstlane(
      blockIdx.x * wavesPerBlk + (threadIdx.x >> 5));

  for (int tile = wave0; tile < nTiles; tile += nWaves) {
    const int row = tile >> TPR_LOG2;                  // scalar
    const int t0  = (tile & TPR_MASK) << 8;            // scalar

    const float* src = chan + (size_t)row * T_LEN;     // scalar base
    float*       dst = outc + (size_t)row * T_LEN;     // scalar base

    // ---- B matrix: 16 consecutive samples per lane (contiguous k order, same as A).
    // K = e + (hi?16:0), N = lane%16  ->  x[t0 + N*16 - 16 + (hi?16:0) + e]
    const int tb = t0 + bOff;
    // Always-in-bounds load (clamped); zero only in the rare halo-below-zero
    // case (lane 0 of each row's first tile) so the hot path is branch-free.
    const v4f* p = (const v4f*)(src + (tb >= 0 ? tb : 0));
    v4f x0 = p[0], x1 = p[1], x2 = p[2], x3 = p[3];
    if (tb < 0) { x0 = (v4f){}; x1 = (v4f){}; x2 = (v4f){}; x3 = (v4f){}; }

    float xs[16];
    *(v4f*)(xs + 0)  = x0;  *(v4f*)(xs + 4)  = x1;
    *(v4f*)(xs + 8)  = x2;  *(v4f*)(xs + 12) = x3;

    v16h bhi, blo;
#pragma unroll
    for (int e = 0; e < 16; ++e) {
      const _Float16 h = (_Float16)xs[e];
      bhi[e] = h;
      blo[e] = (_Float16)(xs[e] - (float)h);
    }

    // ---- Y = Whi*Xhi + Whi*Xlo + Wlo*Xhi  (f32 accumulate; lo*lo ~1e-7, dropped)
    v8f acc = {};
    acc = __builtin_amdgcn_wmma_f32_16x16x32_f16(false, ahi, false, bhi,
                                                 (short)0, acc, false, false);
    acc = __builtin_amdgcn_wmma_f32_16x16x32_f16(false, ahi, false, blo,
                                                 (short)0, acc, false, false);
    acc = __builtin_amdgcn_wmma_f32_16x16x32_f16(false, alo, false, bhi,
                                                 (short)0, acc, false, false);

    // ---- Store: C/D layout -> lane owns 8 consecutive outputs. Non-temporal:
    // the output stream is never re-read; keep it out of L2's way.
    float* o = dst + t0 + oOff;
    v4f s0 = {acc[0], acc[1], acc[2], acc[3]};
    v4f s1 = {acc[4], acc[5], acc[6], acc[7]};
    __builtin_nontemporal_store(s0, (v4f*)o);
    __builtin_nontemporal_store(s1, (v4f*)o + 1);
  }
}

extern "C" void kernel_launch(void* const* d_in, const int* in_sizes, int n_in,
                              void* d_out, int out_size, void* d_ws, size_t ws_size,
                              hipStream_t stream) {
  (void)n_in; (void)out_size; (void)d_ws; (void)ws_size;
  const float* left  = (const float*)d_in[0];
  const float* right = (const float*)d_in[1];
  float* out = (float*)d_out;

  const int R = in_sizes[0] / T_LEN;    // 64 rows per channel

  // 65536 tiles per channel; 8 waves/block, grid.y = channel;
  // grid-stride -> 2 tiles per wave.
  dim3 block(256), grid(4096, 2);
  dispersion_iir_wmma<<<grid, block, 0, stream>>>(left, right, out, R);
}
